// HybridTokenMixer_5686536700036
// MI455X (gfx1250) — compile-verified, matrix operand order checked by reading, not run on hardware
//
#include <hip/hip_runtime.h>
#include <hip/hip_bf16.h>
#include <math.h>

typedef __attribute__((ext_vector_type(16))) _Float16 v16h;
typedef __attribute__((ext_vector_type(8)))  _Float16 v8h;
typedef __attribute__((ext_vector_type(8)))  float    v8f;
typedef __attribute__((ext_vector_type(4)))  int      v4i;

#if __has_builtin(__builtin_amdgcn_global_load_async_to_lds_b128) && __has_builtin(__builtin_amdgcn_s_wait_asynccnt)
#define USE_ASYNC_LDS 1
#endif

typedef __attribute__((address_space(1))) v4i gv4i;   // global int4
typedef __attribute__((address_space(3))) v4i lv4i;   // LDS int4

// ---------------- problem constants ----------------
#define BB 32
#define CC 128
#define DD 64
#define HH 64
#define WW 64
#define HWHW 4096          // H*W
#define INNER 16
// padded HWC f16 buffer for x11: (B, 70, 72, 64)
#define PH 70
#define PW 72

__device__ __forceinline__ float gelu_f(float x) {
    return 0.5f * x * (1.0f + erff(x * 0.70710678118654752f));
}
__device__ __forceinline__ float sigmoid_f(float x) { return 1.0f / (1.0f + expf(-x)); }

__device__ __forceinline__ int kmap(int lane, int h) {
    int base = (lane & 16) ? 8 : 0;
    return base + h + ((h >= 8) ? 8 : 0);
}

// Load a 16x32 (MxK) fragment row: p points at K=0 of this lane's 32-wide K chunk.
// Per ISA 7.12.2 the lane needs halves K=base..base+7 and K=base+16..base+23.
__device__ __forceinline__ v16h load_frag(const _Float16* p, int lane) {
    int base = (lane & 16) ? 8 : 0;
    v8h lo = *(const v8h*)(p + base);
    v8h hi = *(const v8h*)(p + base + 16);
    v16h r;
#pragma unroll
    for (int i = 0; i < 8; ++i) { r[i] = lo[i]; r[i + 8] = hi[i]; }
    return r;
}

__device__ __forceinline__ v8f wmma16(v16h a, v16h b, v8f c) {
    return __builtin_amdgcn_wmma_f32_16x16x32_f16(false, a, false, b, (short)0, c, false, false);
}

__device__ float block_reduce_sum(float v, float* buf) {
    int t = threadIdx.x;
    buf[t] = v; __syncthreads();
    for (int s = blockDim.x >> 1; s > 0; s >>= 1) {
        if (t < s) buf[t] += buf[t + s];
        __syncthreads();
    }
    float r = buf[0]; __syncthreads();
    return r;
}
__device__ float block_reduce_max(float v, float* buf) {
    int t = threadIdx.x;
    buf[t] = v; __syncthreads();
    for (int s = blockDim.x >> 1; s > 0; s >>= 1) {
        if (t < s) buf[t] = fmaxf(buf[t], buf[t + s]);
        __syncthreads();
    }
    float r = buf[0]; __syncthreads();
    return r;
}

// ---------------- small helper kernels ----------------
__global__ void k_zero(float* p, int n) {
    int i = blockIdx.x * blockDim.x + threadIdx.x;
    if (i < n) p[i] = 0.0f;
}

// per-(b,c) reductions: x1 -> 3x3 adaptive pool + mean; x2 -> mean + max
__global__ void k_pool(const float* __restrict__ x, float* pool9, float* x1mean,
                       float* x2mean, float* x2max) {
    __shared__ float red[256];
    int blk = blockIdx.x; int b = blk >> 7; int c = blk & 127;
    const float* in = x + ((size_t)b * CC + c) * HWHW;
    const int HS[3] = {0, 21, 42};
    const int HE[3] = {22, 43, 64};
    if (c < DD) {
        float acc[9] = {0,0,0,0,0,0,0,0,0};
        float tot = 0.0f;
        for (int p = threadIdx.x; p < HWHW; p += 256) {
            int h = p >> 6, w = p & 63;
            float v = in[p]; tot += v;
#pragma unroll
            for (int i = 0; i < 3; ++i) {
                if (h < HS[i] || h >= HE[i]) continue;
#pragma unroll
                for (int j = 0; j < 3; ++j)
                    if (w >= HS[j] && w < HE[j]) acc[i * 3 + j] += v;
            }
        }
        for (int r = 0; r < 9; ++r) {
            float s = block_reduce_sum(acc[r], red);
            if (threadIdx.x == 0) {
                int i = r / 3, j = r % 3;
                float area = (float)((HE[i] - HS[i]) * (HE[j] - HS[j]));
                pool9[((size_t)b * DD + c) * 9 + r] = s / area;
            }
        }
        float s = block_reduce_sum(tot, red);
        if (threadIdx.x == 0) x1mean[b * DD + c] = s / (float)HWHW;
    } else {
        int c2 = c - DD;
        float tot = 0.0f, mx = -3.4e38f;
        for (int p = threadIdx.x; p < HWHW; p += 256) {
            float v = in[p]; tot += v; mx = fmaxf(mx, v);
        }
        float s = block_reduce_sum(tot, red);
        float m = block_reduce_max(mx, red);
        if (threadIdx.x == 0) { x2mean[b * DD + c2] = s / (float)HWHW; x2max[b * DD + c2] = m; }
    }
}

// dynamic projection: both the 3x3-pool path (-> per-(b,c) 3x3 dw weights) and the
// mean path (-> per-(b,c) bias). Single block, 256 threads (tiny GEMMs + batch-BN).
__global__ void k_dynproj(const float* __restrict__ pool9, const float* __restrict__ x1mean,
                          const float* __restrict__ dyn_w, const float* __restrict__ dyn_b,
                          const float* __restrict__ dp1_w, const float* __restrict__ dp1_b,
                          const float* __restrict__ dp1_g, const float* __restrict__ dp1_be,
                          const float* __restrict__ dp2_w, const float* __restrict__ dp2_b,
                          float* wdyn, float* bdyn) {
    __shared__ float z1[BB * 16 * 9];   // 4608
    __shared__ float z1b[BB * 16];      // 512
    __shared__ float sc[16], sh[16], scb[16], shb[16];
    int tid = threadIdx.x;
    for (int i = tid; i < 4608; i += 256) {
        int b = i / 144; int r = i % 144; int c16 = r / 9; int k = r % 9;
        float acc = dp1_b[c16];
        for (int ci = 0; ci < DD; ++ci)
            acc += dp1_w[c16 * DD + ci] * pool9[((size_t)b * DD + ci) * 9 + k];
        z1[i] = acc;
    }
    for (int i = tid; i < 512; i += 256) {
        int b = i / 16; int c16 = i % 16;
        float acc = dp1_b[c16];
        for (int ci = 0; ci < DD; ++ci)
            acc += dp1_w[c16 * DD + ci] * x1mean[b * DD + ci];
        z1b[i] = acc;
    }
    __syncthreads();
    if (tid < 16) {
        float s = 0, q = 0;
        for (int b = 0; b < BB; ++b)
            for (int k = 0; k < 9; ++k) { float v = z1[(b * 16 + tid) * 9 + k]; s += v; q += v * v; }
        float m = s / 288.0f, var = q / 288.0f - m * m;
        float a = dp1_g[tid] * rsqrtf(var + 1e-5f);
        sc[tid] = a; sh[tid] = dp1_be[tid] - m * a;
        s = 0; q = 0;
        for (int b = 0; b < BB; ++b) { float v = z1b[b * 16 + tid]; s += v; q += v * v; }
        m = s / 32.0f; var = q / 32.0f - m * m;
        a = dp1_g[tid] * rsqrtf(var + 1e-5f);
        scb[tid] = a; shb[tid] = dp1_be[tid] - m * a;
    }
    __syncthreads();
    for (int i = tid; i < 4608; i += 256) { int c16 = (i / 9) % 16; z1[i] = gelu_f(z1[i] * sc[c16] + sh[c16]); }
    for (int i = tid; i < 512; i += 256) { int c16 = i % 16; z1b[i] = gelu_f(z1b[i] * scb[c16] + shb[c16]); }
    __syncthreads();
    // second conv + softmax over G=2 + combine with dyn_w / dyn_b
    for (int i = tid; i < BB * DD * 9; i += 256) {
        int b = i / (DD * 9); int r = i % (DD * 9); int c = r / 9; int k = r % 9;
        float s0 = dp2_b[c], s1 = dp2_b[DD + c];
        for (int c16 = 0; c16 < 16; ++c16) {
            float z = z1[(b * 16 + c16) * 9 + k];
            s0 += dp2_w[c * 16 + c16] * z;
            s1 += dp2_w[(DD + c) * 16 + c16] * z;
        }
        float mx = fmaxf(s0, s1);
        float e0 = expf(s0 - mx), e1 = expf(s1 - mx);
        float inv = 1.0f / (e0 + e1);
        wdyn[i] = e0 * inv * dyn_w[c * 9 + k] + e1 * inv * dyn_w[(DD + c) * 9 + k];
    }
    for (int i = tid; i < BB * DD; i += 256) {
        int b = i / DD; int c = i % DD;
        float s0 = dp2_b[c], s1 = dp2_b[DD + c];
        for (int c16 = 0; c16 < 16; ++c16) {
            float z = z1b[b * 16 + c16];
            s0 += dp2_w[c * 16 + c16] * z;
            s1 += dp2_w[(DD + c) * 16 + c16] * z;
        }
        float mx = fmaxf(s0, s1);
        float e0 = expf(s0 - mx), e1 = expf(s1 - mx);
        float inv = 1.0f / (e0 + e1);
        bdyn[i] = e0 * inv * dyn_b[c] + e1 * inv * dyn_b[DD + c];
    }
}

// channel attention: sigmoid(w2@relu(w1@mean) + w2@relu(w1@max))
__global__ void k_ca(const float* __restrict__ x2mean, const float* __restrict__ x2max,
                     const float* __restrict__ w1, const float* __restrict__ w2, float* att_c) {
    __shared__ float t[8];
    int b = blockIdx.x;
    if (threadIdx.x < 8) {
        int j = threadIdx.x;
        float a1 = 0.0f, a2 = 0.0f;
        for (int i = 0; i < DD; ++i) {
            a1 += w1[j * DD + i] * x2mean[b * DD + i];
            a2 += w1[j * DD + i] * x2max[b * DD + i];
        }
        t[j] = fmaxf(a1, 0.0f) + fmaxf(a2, 0.0f);
    }
    __syncthreads();
    int c = threadIdx.x;
    if (c < DD) {
        float o = 0.0f;
        for (int j = 0; j < 8; ++j) o += w2[c * 8 + j] * t[j];
        att_c[b * DD + c] = sigmoid_f(o);
    }
}

// combined 7x7 weight (c7 + padded c5 + padded c3)
__device__ __forceinline__ float comb_w(const float* c3w, const float* c5w, const float* c7w,
                                        int n, int k, int r, int s) {
    float v = c7w[((n * DD + k) * 7 + r) * 7 + s];
    if (r >= 1 && r < 6 && s >= 1 && s < 6) v += c5w[((n * DD + k) * 5 + (r - 1)) * 5 + (s - 1)];
    if (r >= 2 && r < 5 && s >= 2 && s < 5) v += c3w[((n * DD + k) * 3 + (r - 2)) * 3 + (s - 2)];
    return v;
}

// build all f16 WMMA B-fragment buffers (per-lane contiguous layout) + combined bias
__global__ void k_prep(const float* __restrict__ down_w, const float* __restrict__ up_w,
                       const float* __restrict__ pw1_w, const float* __restrict__ pw2_w,
                       const float* __restrict__ c3w, const float* __restrict__ c3b,
                       const float* __restrict__ c5w, const float* __restrict__ c5b,
                       const float* __restrict__ c7w, const float* __restrict__ c7b,
                       _Float16* fdown, _Float16* fup, _Float16* fpw1, _Float16* fpw2,
                       _Float16* fc7, float* cbias) {
    int i = blockIdx.x * 256 + threadIdx.x;
    if (i < 4096) {   // down/up: Cout=64 (nt 0..3), KB=2, idx=((nt*2+kb)*32+lane)*16+h
        int h = i & 15, lane = (i >> 4) & 31, kb = (i >> 9) & 1, nt = i >> 10;
        int n = nt * 16 + (lane & 15); int k = kb * 32 + kmap(lane, h);
        fdown[i] = (_Float16)down_w[n * DD + k];
        fup[i]   = (_Float16)up_w[n * DD + k];
    }
    if (i < 2048) {   // pw1: Cout=16 (nt=0), KB=4
        int h = i & 15, lane = (i >> 4) & 31, kb = i >> 9;
        int n = lane & 15; int k = kb * 32 + kmap(lane, h);
        fpw1[i] = (_Float16)pw1_w[n * CC + k];
    }
    if (i < 4096) {   // pw2: Cout=128 (nt 0..7), KB=1, K padded 16->32
        int h = i & 15, lane = (i >> 4) & 31, nt = i >> 9;
        int n = nt * 16 + (lane & 15); int k = kmap(lane, h);
        fpw2[i] = (_Float16)((k < 16) ? pw2_w[n * 16 + k] : 0.0f);
    }
    if (i < 200704) { // conv7: idx=(((nt*49+tap)*2+kb)*32+lane)*16+h
        int h = i & 15, lane = (i >> 4) & 31, kb = (i >> 9) & 1;
        int q = i >> 10; int tap = q % 49, nt = q / 49;
        int n = nt * 16 + (lane & 15); int k = kb * 32 + kmap(lane, h);
        int r = tap / 7, s = tap % 7;
        fc7[i] = (_Float16)comb_w(c3w, c5w, c7w, n, k, r, s);
    }
    if (i < DD) cbias[i] = c3b[i] + c5b[i] + c7b[i];
}

// dynamic depthwise 3x3 (y1) -> xcat channels [0,64)
__global__ void k_dyndw(const float* __restrict__ x, const float* __restrict__ wdyn,
                        const float* __restrict__ bdyn, float* __restrict__ xcat) {
    __shared__ float wk[9]; __shared__ float bsh;
    int blk = blockIdx.x; int b = blk >> 6; int c = blk & 63;
    if (threadIdx.x < 9) wk[threadIdx.x] = wdyn[(b * DD + c) * 9 + threadIdx.x];
    if (threadIdx.x == 9) bsh = bdyn[b * DD + c];
    __syncthreads();
    const float* in = x + ((size_t)b * CC + c) * HWHW;
    float* out = xcat + ((size_t)b * CC + c) * HWHW;
    for (int p = threadIdx.x; p < HWHW; p += blockDim.x) {
        int h = p >> 6, w = p & 63;
        float acc = bsh;
#pragma unroll
        for (int r = 0; r < 3; ++r) {
            int gh = h + r - 1; if (gh < 0 || gh >= HH) continue;
#pragma unroll
            for (int s = 0; s < 3; ++s) {
                int gw = w + s - 1; if (gw < 0 || gw >= WW) continue;
                acc += wk[r * 3 + s] * in[gh * WW + gw];
            }
        }
        out[p] = acc;
    }
}

// ---------------- WMMA pointwise GEMM (1x1 conv) ----------------
// out[b, out_coff+ch, p] = bias[ch] + sum_k W[ch,k] * affine(in[b, in_coff+k, p])
// optional output GELU, per-channel BN-stat accumulation, and optional f16
// padded-HWC output (for feeding the async-staged 7x7 conv).
__global__ void k_pw_wmma(const float* __restrict__ in, const _Float16* __restrict__ Wf,
                          const float* __restrict__ bias, float* __restrict__ out,
                          _Float16* __restrict__ out_hwc,
                          int CinP, int CinReal, int Cout,
                          int in_ctot, int in_coff, int out_ctot, int out_coff,
                          const float* __restrict__ in_scale, const float* __restrict__ in_shift,
                          int do_gelu, float* __restrict__ stats) {
    __shared__ _Float16 As[16 * 128];
    __shared__ float ssum[128];
    __shared__ float ssq[128];
    int tpix = blockIdx.x * 16;
    int b = tpix / HWHW;
    int p0 = tpix % HWHW;
    if (stats) {
        if ((int)threadIdx.x < Cout) { ssum[threadIdx.x] = 0.0f; ssq[threadIdx.x] = 0.0f; }
    }
    for (int idx = threadIdx.x; idx < 16 * CinP; idx += blockDim.x) {
        int c = idx >> 4; int xp = idx & 15;
        float v = 0.0f;
        if (c < CinReal) {
            v = in[((size_t)b * in_ctot + in_coff + c) * HWHW + p0 + xp];
            if (in_scale) v = v * in_scale[c] + in_shift[c];
        }
        As[xp * CinP + c] = (_Float16)v;
    }
    __syncthreads();
    int lane = threadIdx.x & 31, nt = threadIdx.x >> 5;
    int mrow = lane & 15;
    int KB = CinP >> 5;
    v8f acc = {};
    for (int kb = 0; kb < KB; ++kb) {
        v16h a = load_frag(&As[mrow * CinP + kb * 32], lane);
        v16h bw = *(const v16h*)(Wf + (((size_t)(nt * KB + kb)) * 32 + lane) * 16);
        acc = wmma16(a, bw, acc);
    }
    int n = lane & 15; int ch = nt * 16 + n;
    float bb = bias[ch];
    float ls = 0.0f, lq = 0.0f;
#pragma unroll
    for (int v = 0; v < 8; ++v) {
        int m = v + ((lane >> 4) << 3);
        float val = acc[v] + bb;
        if (do_gelu) val = gelu_f(val);
        if (out_hwc) {
            int p = p0 + m; int h = p >> 6, w = p & 63;
            out_hwc[((size_t)(b * PH + h + 3) * PW + (w + 3)) * DD + ch] = (_Float16)val;
        } else {
            out[((size_t)b * out_ctot + out_coff + ch) * HWHW + p0 + m] = val;
        }
        ls += val; lq += val * val;
    }
    if (stats) {
        atomicAdd(&ssum[ch], ls);
        atomicAdd(&ssq[ch], lq);
        __syncthreads();
        if ((int)threadIdx.x < Cout) {
            atomicAdd(&stats[threadIdx.x], ssum[threadIdx.x]);
            atomicAdd(&stats[Cout + threadIdx.x], ssq[threadIdx.x]);
        }
    }
}

// ---------------- WMMA 7x7 implicit-GEMM conv (combined 3/5/7 kernel) ----------------
// Input: padded HWC f16 (B,70,72,64). Block = 32 pixels of one row, 8 waves:
// 2 M-tiles x 4 N-tiles. Halo staged LDS [r][x][c], async copy when available.
__global__ void k_conv7_wmma(const _Float16* __restrict__ x11p, const _Float16* __restrict__ Wf,
                             const float* __restrict__ bias, float* __restrict__ xs) {
    __shared__ _Float16 tile[7 * 38 * 64];   // 34 KB
    int t = blockIdx.x;             // 4096 tiles
    int b = t >> 7;                 // 128 tiles per image
    int rem = t & 127;
    int h = rem >> 1;
    int w0 = (rem & 1) * 32;
    const _Float16* gbase = x11p + ((size_t)(b * PH + h) * PW + w0) * DD;
    __builtin_prefetch(gbase, 0, 1);
    // 7 rows x (38 pixels x 64 ch) contiguous spans; 2128 x 16B chunks
#ifdef USE_ASYNC_LDS
    for (int idx = threadIdx.x; idx < 2128; idx += 256) {
        int r = idx / 304, o = idx - r * 304;
        const _Float16* g = gbase + (size_t)r * (PW * DD) + o * 8;
        _Float16* l = &tile[r * (38 * DD) + o * 8];
        __builtin_amdgcn_global_load_async_to_lds_b128((gv4i*)g, (lv4i*)l, 0, 0);
    }
    __builtin_amdgcn_s_wait_asynccnt(0);
    __syncthreads();
#else
    for (int idx = threadIdx.x; idx < 2128; idx += 256) {
        int r = idx / 304, o = idx - r * 304;
        *(v8h*)&tile[r * (38 * DD) + o * 8] =
            *(const v8h*)(gbase + (size_t)r * (PW * DD) + o * 8);
    }
    __syncthreads();
#endif
    int lane = threadIdx.x & 31;
    int wv = threadIdx.x >> 5;      // 0..7
    int mt = wv >> 2;               // which 16-pixel half of the 32-pixel tile
    int nt = wv & 3;                // 16-channel N tile
    int mrow = lane & 15;
    v8f acc = {};
    for (int tap = 0; tap < 49; ++tap) {
        int r = tap / 7, s = tap % 7;
        const _Float16* arow = &tile[((r * 38) + mt * 16 + mrow + s) * DD];
#pragma unroll
        for (int kb = 0; kb < 2; ++kb) {
            v16h a = load_frag(arow + kb * 32, lane);
            v16h bw = *(const v16h*)(Wf + ((((size_t)nt * 49 + tap) * 2 + kb) * 32 + lane) * 16);
            acc = wmma16(a, bw, acc);
        }
    }
    int n = lane & 15; int ch = nt * 16 + n;
    float bb = bias[ch];
#pragma unroll
    for (int v = 0; v < 8; ++v) {
        int m = v + ((lane >> 4) << 3);
        xs[((size_t)b * DD + ch) * HWHW + h * WW + w0 + mt * 16 + m] = acc[v] + bb;
    }
}

// spatial attention: channel mean/max of xs
__global__ void k_sa_reduce(const float* __restrict__ xs, float* samean, float* samax) {
    int i = blockIdx.x * blockDim.x + threadIdx.x;
    if (i >= BB * HWHW) return;
    int b = i / HWHW, p = i % HWHW;
    float s = 0.0f, mx = -3.4e38f;
    for (int c = 0; c < DD; ++c) {
        float v = xs[((size_t)b * DD + c) * HWHW + p];
        s += v; mx = fmaxf(mx, v);
    }
    samean[i] = s / (float)DD; samax[i] = mx;
}

// 7x7 conv over [mean,max] channels + sigmoid
__global__ void k_sa_conv(const float* __restrict__ samean, const float* __restrict__ samax,
                          const float* __restrict__ saw, float* att_s) {
    int i = blockIdx.x * blockDim.x + threadIdx.x;
    if (i >= BB * HWHW) return;
    int b = i / HWHW, p = i % HWHW;
    int h = p >> 6, w = p & 63;
    float acc = 0.0f;
    for (int r = 0; r < 7; ++r) {
        int gh = h + r - 3; if (gh < 0 || gh >= HH) continue;
        for (int s = 0; s < 7; ++s) {
            int gw = w + s - 3; if (gw < 0 || gw >= WW) continue;
            int np = b * HWHW + gh * WW + gw;
            acc += samean[np] * saw[r * 7 + s] + samax[np] * saw[49 + r * 7 + s];
        }
    }
    att_s[i] = sigmoid_f(acc);
}

// xs = xs*att_s + x2*att_c   (in place)
__global__ void k_combine(float* __restrict__ xs, const float* __restrict__ x,
                          const float* __restrict__ att_s, const float* __restrict__ att_c) {
    size_t i = (size_t)blockIdx.x * blockDim.x + threadIdx.x;
    if (i >= (size_t)BB * DD * HWHW) return;
    int b = (int)(i / ((size_t)DD * HWHW));
    int c = (int)((i / HWHW) % DD);
    int p = (int)(i % HWHW);
    xs[i] = xs[i] * att_s[b * HWHW + p]
          + x[((size_t)b * CC + DD + c) * HWHW + p] * att_c[b * DD + c];
}

// static depthwise 3x3 + GELU, with BN1 stat accumulation
__global__ void k_dw_stats(const float* __restrict__ xcat, const float* __restrict__ dww,
                           const float* __restrict__ dwb, float* __restrict__ u0,
                           float* __restrict__ stats) {
    __shared__ float wk[9]; __shared__ float red[256];
    int blk = blockIdx.x; int b = blk >> 7; int c = blk & 127;
    if (threadIdx.x < 9) wk[threadIdx.x] = dww[c * 9 + threadIdx.x];
    __syncthreads();
    const float* in = xcat + ((size_t)b * CC + c) * HWHW;
    float* out = u0 + ((size_t)b * CC + c) * HWHW;
    float bb = dwb[c];
    float ls = 0.0f, lq = 0.0f;
    for (int p = threadIdx.x; p < HWHW; p += 256) {
        int h = p >> 6, w = p & 63;
        float acc = bb;
#pragma unroll
        for (int r = 0; r < 3; ++r) {
            int gh = h + r - 1; if (gh < 0 || gh >= HH) continue;
#pragma unroll
            for (int s = 0; s < 3; ++s) {
                int gw = w + s - 1; if (gw < 0 || gw >= WW) continue;
                acc += wk[r * 3 + s] * in[gh * WW + gw];
            }
        }
        float g = gelu_f(acc);
        out[p] = g; ls += g; lq += g * g;
    }
    float ts = block_reduce_sum(ls, red);
    float tq = block_reduce_sum(lq, red);
    if (threadIdx.x == 0) { atomicAdd(&stats[c], ts); atomicAdd(&stats[CC + c], tq); }
}

__global__ void k_finalize_bn(const float* __restrict__ stats, const float* __restrict__ g,
                              const float* __restrict__ be, float* scale, float* shift,
                              int Cn, float N) {
    int c = blockIdx.x * blockDim.x + threadIdx.x;
    if (c < Cn) {
        float m = stats[c] / N;
        float var = stats[Cn + c] / N - m * m;
        float a = g[c] * rsqrtf(var + 1e-5f);
        scale[c] = a; shift[c] = be[c] - m * a;
    }
}

__global__ void k_final(const float* __restrict__ u2, const float* __restrict__ xcat,
                        const float* __restrict__ scale, const float* __restrict__ shift,
                        float* __restrict__ out) {
    size_t i = (size_t)blockIdx.x * blockDim.x + threadIdx.x;
    if (i >= (size_t)BB * CC * HWHW) return;
    int c = (int)((i / HWHW) % CC);
    out[i] = u2[i] * scale[c] + shift[c] + xcat[i];
}

// ---------------- host side ----------------
extern "C" void kernel_launch(void* const* d_in, const int* in_sizes, int n_in,
                              void* d_out, int out_size, void* d_ws, size_t ws_size,
                              hipStream_t stream) {
    (void)in_sizes; (void)n_in; (void)out_size; (void)ws_size;
    const float* x      = (const float*)d_in[0];
    const float* dyn_w  = (const float*)d_in[1];
    const float* dyn_b  = (const float*)d_in[2];
    const float* dp1_w  = (const float*)d_in[3];
    const float* dp1_b  = (const float*)d_in[4];
    const float* dp1_g  = (const float*)d_in[5];
    const float* dp1_be = (const float*)d_in[6];
    const float* dp2_w  = (const float*)d_in[7];
    const float* dp2_b  = (const float*)d_in[8];
    const float* ca_w1  = (const float*)d_in[9];
    const float* ca_w2  = (const float*)d_in[10];
    const float* down_w = (const float*)d_in[11];
    const float* down_b = (const float*)d_in[12];
    const float* c3_w   = (const float*)d_in[13];
    const float* c3_b   = (const float*)d_in[14];
    const float* c5_w   = (const float*)d_in[15];
    const float* c5_b   = (const float*)d_in[16];
    const float* c7_w   = (const float*)d_in[17];
    const float* c7_b   = (const float*)d_in[18];
    const float* sa_w   = (const float*)d_in[19];
    const float* up_w   = (const float*)d_in[20];
    const float* up_b   = (const float*)d_in[21];
    const float* dw_w   = (const float*)d_in[22];
    const float* dw_b   = (const float*)d_in[23];
    const float* bn1_g  = (const float*)d_in[24];
    const float* bn1_b  = (const float*)d_in[25];
    const float* pw1_w  = (const float*)d_in[26];
    const float* pw1_b  = (const float*)d_in[27];
    const float* bn2_g  = (const float*)d_in[28];
    const float* bn2_b  = (const float*)d_in[29];
    const float* pw2_w  = (const float*)d_in[30];
    const float* pw2_b  = (const float*)d_in[31];
    const float* bn3_g  = (const float*)d_in[32];
    const float* bn3_b  = (const float*)d_in[33];
    float* out = (float*)d_out;

    float* w = (float*)d_ws;
    // workspace layout (floats)
    const size_t OFF_POOL9  = 0;             // 18432
    const size_t OFF_X1MEAN = 18432;         // 2048
    const size_t OFF_X2MEAN = 20480;         // 2048
    const size_t OFF_X2MAX  = 22528;         // 2048
    const size_t OFF_WDYN   = 24576;         // 18432
    const size_t OFF_BDYN   = 43008;         // 2048
    const size_t OFF_ATTC   = 45056;         // 2048
    const size_t OFF_STATS  = 47104;         // bn1:256, bn2:32, bn3:256 -> 544
    const size_t OFF_SCALE  = 47648;         // sc1/sh1 (256), sc2/sh2 (32), sc3/sh3 (256)
    const size_t OFF_CBIAS  = 48192;         // 64
    const size_t OFF_SAMEAN = 48256;         // 131072
    const size_t OFF_SAMAX  = 179328;        // 131072
    const size_t OFF_ATTS   = 310400;        // 131072
    const size_t OFF_FRAG   = 441472;        // 215040 halves = 107520 floats
    const size_t OFF_X11P   = 548992;        // 10321920 halves = 5160960 fl (round 5161216); u1 aliases
    const size_t OFF_XS     = OFF_X11P + 5161216;      // 8388608
    const size_t OFF_XCAT   = OFF_XS + 8388608;        // 16777216
    const size_t OFF_U0     = OFF_XCAT + 16777216;     // 16777216 (aliased by u2)

    float* pool9  = w + OFF_POOL9;
    float* x1mean = w + OFF_X1MEAN;
    float* x2mean = w + OFF_X2MEAN;
    float* x2max  = w + OFF_X2MAX;
    float* wdyn   = w + OFF_WDYN;
    float* bdyn   = w + OFF_BDYN;
    float* att_c  = w + OFF_ATTC;
    float* stats1 = w + OFF_STATS;          // 256
    float* stats2 = w + OFF_STATS + 256;    // 32
    float* stats3 = w + OFF_STATS + 288;    // 256
    float* sc1 = w + OFF_SCALE;       float* sh1 = sc1 + 128;
    float* sc2 = sc1 + 256;           float* sh2 = sc2 + 16;
    float* sc3 = sc2 + 32;            float* sh3 = sc3 + 128;
    float* cbias  = w + OFF_CBIAS;
    float* samean = w + OFF_SAMEAN;
    float* samax  = w + OFF_SAMAX;
    float* att_s  = w + OFF_ATTS;
    _Float16* frag  = (_Float16*)(w + OFF_FRAG);
    _Float16* fdown = frag;             // 4096
    _Float16* fup   = frag + 4096;      // 4096
    _Float16* fpw1  = frag + 8192;      // 2048
    _Float16* fpw2  = frag + 10240;     // 4096
    _Float16* fc7   = frag + 14336;     // 200704
    _Float16* x11p = (_Float16*)(w + OFF_X11P);
    float* xs   = w + OFF_XS;
    float* xcat = w + OFF_XCAT;
    float* u0   = w + OFF_U0;
    float* u1   = w + OFF_X11P;   // x11p dead after conv7
    float* u2   = u0;             // u0 dead after pw1

    const float Nbn = (float)(BB * HWHW);

    // 1. zero stats + zero padded x11p (border halo must be 0)
    k_zero<<<3, 256, 0, stream>>>(w + OFF_STATS, 544);
    k_zero<<<20160, 256, 0, stream>>>(w + OFF_X11P, 5160960);
    // 2. pooled reductions
    k_pool<<<BB * CC, 256, 0, stream>>>(x, pool9, x1mean, x2mean, x2max);
    // 3. dynamic projection (w, bias)
    k_dynproj<<<1, 256, 0, stream>>>(pool9, x1mean, dyn_w, dyn_b, dp1_w, dp1_b, dp1_g, dp1_be,
                                     dp2_w, dp2_b, wdyn, bdyn);
    // 4. channel attention
    k_ca<<<BB, 64, 0, stream>>>(x2mean, x2max, ca_w1, ca_w2, att_c);
    // 5. weight prep (f16 fragment swizzle + combined 7x7 + combined bias)
    k_prep<<<784, 256, 0, stream>>>(down_w, up_w, pw1_w, pw2_w, c3_w, c3_b, c5_w, c5_b,
                                    c7_w, c7_b, fdown, fup, fpw1, fpw2, fc7, cbias);
    // 6. y1: dynamic depthwise 3x3 -> xcat[:, 0:64]
    k_dyndw<<<BB * DD, 256, 0, stream>>>(x, wdyn, bdyn, xcat);
    // 7. x11p = down(x2) as padded-HWC f16     [WMMA]
    k_pw_wmma<<<8192, 128, 0, stream>>>(x, fdown, down_b, nullptr, x11p, 64, 64, 64,
                                        CC, DD, DD, 0, nullptr, nullptr, 0, nullptr);
    // 8. xs = combined 7x7 conv(x11p)   [WMMA + async-to-LDS staging]
    k_conv7_wmma<<<4096, 256, 0, stream>>>(x11p, fc7, cbias, xs);
    // 9/10. spatial attention
    k_sa_reduce<<<512, 256, 0, stream>>>(xs, samean, samax);
    k_sa_conv<<<512, 256, 0, stream>>>(samean, samax, sa_w, att_s);
    // 11. xs = xs*att_s + x2*att_c
    k_combine<<<32768, 256, 0, stream>>>(xs, x, att_s, att_c);
    // 12. y2 = up(xs) -> xcat[:, 64:128]   [WMMA]
    k_pw_wmma<<<8192, 128, 0, stream>>>(xs, fup, up_b, xcat, nullptr, 64, 64, 64,
                                        DD, 0, CC, DD, nullptr, nullptr, 0, nullptr);
    // 13. u0 = gelu(dw3x3(xcat)) + bn1 stats
    k_dw_stats<<<BB * CC, 256, 0, stream>>>(xcat, dw_w, dw_b, u0, stats1);
    // 14. bn1 scale/shift
    k_finalize_bn<<<1, 128, 0, stream>>>(stats1, bn1_g, bn1_b, sc1, sh1, CC, Nbn);
    // 15. u1 = gelu(pw1(bn1(u0))) + bn2 stats   [WMMA]
    k_pw_wmma<<<8192, 32, 0, stream>>>(u0, fpw1, pw1_b, u1, nullptr, 128, 128, 16,
                                       CC, 0, INNER, 0, sc1, sh1, 1, stats2);
    // 16. bn2
    k_finalize_bn<<<1, 128, 0, stream>>>(stats2, bn2_g, bn2_b, sc2, sh2, INNER, Nbn);
    // 17. u2 = pw2(bn2(u1)) + bn3 stats   [WMMA, K padded 16->32]
    k_pw_wmma<<<8192, 256, 0, stream>>>(u1, fpw2, pw2_b, u2, nullptr, 32, 16, 128,
                                        INNER, 0, CC, 0, sc2, sh2, 0, stats3);
    // 18. bn3
    k_finalize_bn<<<1, 128, 0, stream>>>(stats3, bn3_g, bn3_b, sc3, sh3, CC, Nbn);
    // 19. out = bn3(u2) + xcat
    k_final<<<65536, 256, 0, stream>>>(u2, xcat, sc3, sh3, out);
}
